// GCN_77249281786392
// MI455X (gfx1250) — matrix operand order repaired and checked by reading, not compiled, not run on hardware
//
#include <hip/hip_runtime.h>

typedef __bf16 bf16;
typedef __attribute__((ext_vector_type(16))) __bf16 v16bf;
typedef __attribute__((ext_vector_type(8)))  __bf16 v8bf;
typedef __attribute__((ext_vector_type(8)))  float  v8f;

// pack two floats into two bf16 halves with a single v_cvt_pk_bf16_f32
__device__ __forceinline__ unsigned pack2bf16(float a, float b) {
    unsigned r;
    asm("v_cvt_pk_bf16_f32 %0, %1, %2" : "=v"(r) : "v"(a), "v"(b));
    return r;
}

// ---------------------------------------------------------------------------
// degree / normalization helpers
// ---------------------------------------------------------------------------
__global__ void k_fill(float* __restrict__ p, float v, int n) {
    int i = blockIdx.x * blockDim.x + threadIdx.x;
    if (i < n) p[i] = v;
}

__global__ void k_deg_accum(const int* __restrict__ dst, float* __restrict__ deg, int nE) {
    int i = blockIdx.x * blockDim.x + threadIdx.x;
    if (i < nE) atomicAdd(&deg[dst[i]], 1.0f);
}

__global__ void k_rsqrt(float* __restrict__ p, int n) {
    int i = blockIdx.x * blockDim.x + threadIdx.x;
    if (i < n) p[i] = rsqrtf(p[i]);   // deg >= 1 always (self loops)
}

// ---------------------------------------------------------------------------
// WMMA GEMM: C[M x BN] = A[M x K] * B[K x BN], fp32 in/out, bf16 compute.
// BM=128, BK=32, 256 threads = 8 waves (4 x 2). Wave owns 2 x (BN/32) tiles.
// Double-buffered LDS; branchless clamped staging in K-pairs; packed bf16
// conversion (v_cvt_pk_bf16_f32) + ds_store_b32 commits; NT loads on A.
// ---------------------------------------------------------------------------
template <int BN>
__global__ __launch_bounds__(256)
void k_gemm_bf16(const float* __restrict__ A, const float* __restrict__ Bw,
                 float* __restrict__ C, int M, int K) {
    constexpr int BM  = 128;
    constexpr int BK  = 32;
    constexpr int WTM = 2;                     // 16-row tiles per wave
    constexpr int WTN = BN / 32;               // 16-col tiles per wave
    constexpr int AP  = (BM * BK / 2) / 256;   // A pairs per thread (8)
    constexpr int BP  = (BK * BN / 2) / 256;   // B pairs per thread (8 | 4)
    constexpr int BRS = 256 / (BN / 2);        // B row step (4 | 8)

    __shared__ __align__(16) bf16 As[2][BM][BK + 8];
    __shared__ __align__(16) bf16 Bs[2][BK][BN + 8];

    const int tid  = threadIdx.x;
    const int lane = tid & 31;
    const int wv   = tid >> 5;
    const int wm   = wv & 3;    // 0..3  (M direction)
    const int wn   = wv >> 2;   // 0..1  (N direction)
    const int rowBase = blockIdx.x * BM;

    const int halfSel = lane >> 4;       // 0 | 1
    const int abase   = halfSel * 8;     // A-fragment K offset per ISA layout
    const int mrow    = lane & 15;

    const int a_r = tid >> 4;            // A staging row base (0..15, step 16)
    const int a_c = (tid & 15) * 2;      // A staging col (pair)
    const int b_r = tid / (BN / 2);      // B staging row base
    const int b_c = (tid % (BN / 2)) * 2;// B staging col (pair)

    v8f acc[WTM][WTN];
    const v8f vzero = {0.f, 0.f, 0.f, 0.f, 0.f, 0.f, 0.f, 0.f};
#pragma unroll
    for (int i = 0; i < WTM; ++i)
#pragma unroll
        for (int j = 0; j < WTN; ++j) acc[i][j] = vzero;

    const int nk = (K + BK - 1) / BK;
    float areg[AP][2];
    float breg[BP][2];

    // ---- branchless clamped fetch of one K-tile into registers ----
    auto fetch = [&](int k0) {
#pragma unroll
        for (int i = 0; i < AP; ++i) {
            int gr  = rowBase + a_r + i * 16;
            int gc0 = k0 + a_c, gc1 = gc0 + 1;
            int grc  = gr  < M ? gr  : (M - 1);
            int gc0c = gc0 < K ? gc0 : (K - 1);
            int gc1c = gc1 < K ? gc1 : (K - 1);
            const float* rp = A + (size_t)grc * K;
            float v0 = __builtin_nontemporal_load(rp + gc0c);
            float v1 = __builtin_nontemporal_load(rp + gc1c);
            areg[i][0] = (gr < M && gc0 < K) ? v0 : 0.0f;
            areg[i][1] = (gr < M && gc1 < K) ? v1 : 0.0f;
        }
#pragma unroll
        for (int i = 0; i < BP; ++i) {
            int gk  = k0 + b_r + i * BRS;
            int gkc = gk < K ? gk : (K - 1);
            float2 w = *(const float2*)(Bw + (size_t)gkc * BN + b_c);
            breg[i][0] = (gk < K) ? w.x : 0.0f;
            breg[i][1] = (gk < K) ? w.y : 0.0f;
        }
    };
    // ---- packed bf16 commit of prefetched tile to LDS buffer `buf` ----
    auto commit = [&](int buf) {
#pragma unroll
        for (int i = 0; i < AP; ++i)
            *(unsigned*)&As[buf][a_r + i * 16][a_c] = pack2bf16(areg[i][0], areg[i][1]);
#pragma unroll
        for (int i = 0; i < BP; ++i)
            *(unsigned*)&Bs[buf][b_r + i * BRS][b_c] = pack2bf16(breg[i][0], breg[i][1]);
    };

    fetch(0);
    commit(0);
    __syncthreads();

    for (int kt = 0; kt < nk; ++kt) {
        const int cur  = kt & 1;
        const bool more = (kt + 1) < nk;

        if (more) fetch((kt + 1) * BK);   // overlaps WMMA below

        // ---- A fragments (ISA 7.12.2: lane<16 -> K {0..7,16..23}) ----
        v16bf afrag[WTM];
#pragma unroll
        for (int i = 0; i < WTM; ++i) {
            int r = (wm * WTM + i) * 16 + mrow;
            v8bf lo = *(const v8bf*)&As[cur][r][abase];
            v8bf hi = *(const v8bf*)&As[cur][r][abase + 16];
            afrag[i] = __builtin_shufflevector(lo, hi, 0, 1, 2, 3, 4, 5, 6, 7,
                                               8, 9, 10, 11, 12, 13, 14, 15);
        }
        // ---- B fragments (lane&15 = N col, K {0..7,16..23} / {8..15,24..31}) + WMMA ----
#pragma unroll
        for (int j = 0; j < WTN; ++j) {
            int nb = (wn * WTN + j) * 16;
            v16bf bfrag;
#pragma unroll
            for (int s = 0; s < 8; ++s) {
                bfrag[s]     = Bs[cur][abase + s][nb + mrow];
                bfrag[s + 8] = Bs[cur][abase + 16 + s][nb + mrow];
            }
#pragma unroll
            for (int i = 0; i < WTM; ++i) {
                acc[i][j] = __builtin_amdgcn_wmma_f32_16x16x32_bf16(
                    false, afrag[i], false, bfrag, (short)0, acc[i][j], false, false);
            }
        }

        if (more) commit(1 - cur);
        __syncthreads();
    }

    // ---- store C (VGPR r -> M = r + 8*halfSel, lane&15 -> N) ----
#pragma unroll
    for (int i = 0; i < WTM; ++i) {
#pragma unroll
        for (int j = 0; j < WTN; ++j) {
            int nb = (wn * WTN + j) * 16 + (lane & 15);
#pragma unroll
            for (int r = 0; r < 8; ++r) {
                int m  = (wm * WTM + i) * 16 + r + 8 * halfSel;
                int gr = rowBase + m;
                if (gr < M) C[(size_t)gr * BN + nb] = acc[i][j][r];
            }
        }
    }
}

// ---------------------------------------------------------------------------
// aggregation
// ---------------------------------------------------------------------------
template <int D>
__global__ void k_self_init(const float* __restrict__ h, const float* __restrict__ dinv,
                            float* __restrict__ out, int N) {
    int idx = blockIdx.x * blockDim.x + threadIdx.x;
    if (idx < N * D) {
        int n = idx / D;
        float di = dinv[n];
        out[idx] = h[idx] * di * di;    // self-loop term, fully initializes out
    }
}

template <int D>
__global__ void k_scatter(const float* __restrict__ h, const int* __restrict__ src,
                          const int* __restrict__ dst, const float* __restrict__ dinv,
                          float* __restrict__ out, int nE) {
    constexpr int Q = D / 4;
    int idx = blockIdx.x * blockDim.x + threadIdx.x;
    int total = nE * Q;
    if (idx >= total) return;
    int e = idx / Q, q = idx % Q;
    int s = src[e], d = dst[e];
    float nrm = dinv[s] * dinv[d];
    float4 v  = *(const float4*)(h + (size_t)s * D + q * 4);
    float* op = out + (size_t)d * D + q * 4;
    atomicAdd(op + 0, v.x * nrm);
    atomicAdd(op + 1, v.y * nrm);
    atomicAdd(op + 2, v.z * nrm);
    atomicAdd(op + 3, v.w * nrm);
}

template <int D, bool RELU>
__global__ void k_bias_act(float* __restrict__ z, const float* __restrict__ b, int N) {
    int idx = blockIdx.x * blockDim.x + threadIdx.x;
    if (idx < N * D) {
        float v = z[idx] + b[idx & (D - 1)];
        if (RELU) v = fmaxf(v, 0.0f);
        z[idx] = v;
    }
}

// ---------------------------------------------------------------------------
// link decode: out[e] = dot(z[s[e]], z[t[e]]) over D dims
// ---------------------------------------------------------------------------
__global__ void k_decode(const float* __restrict__ z, const int* __restrict__ s,
                         const int* __restrict__ t, float* __restrict__ out,
                         int nE, int D) {
    int e = blockIdx.x * blockDim.x + threadIdx.x;
    if (e >= nE) return;
    const float4* a = (const float4*)(z + (size_t)s[e] * D);
    const float4* b = (const float4*)(z + (size_t)t[e] * D);
    float acc = 0.0f;
    for (int q = 0; q < D / 4; ++q) {
        float4 x = a[q], y = b[q];
        acc += x.x * y.x + x.y * y.y + x.z * y.z + x.w * y.w;
    }
    out[e] = acc;
}

// ---------------------------------------------------------------------------
extern "C" void kernel_launch(void* const* d_in, const int* in_sizes, int n_in,
                              void* d_out, int out_size, void* d_ws, size_t ws_size,
                              hipStream_t stream) {
    const float* x  = (const float*)d_in[0];
    const float* W1 = (const float*)d_in[1];
    const float* b1 = (const float*)d_in[2];
    const float* W2 = (const float*)d_in[3];
    const float* b2 = (const float*)d_in[4];
    const int* eidx = (const int*)d_in[5];
    const int* lidx = (const int*)d_in[6];

    const int DHID = in_sizes[2];          // 128
    const int DIN  = in_sizes[1] / DHID;   // 3703
    const int DOUT = in_sizes[4];          // 64
    const int N    = in_sizes[0] / DIN;    // 50000
    const int nE   = in_sizes[5] / 2;      // 1,600,000
    const int nL   = in_sizes[6] / 2;      // 200,000

    const int* src = eidx;
    const int* dst = eidx + nE;
    const int* ls  = lidx;
    const int* lt  = lidx + nL;

    char* ws = (char*)d_ws;
    float* dinv = (float*)ws;                                       // N floats
    size_t off1 = ((size_t)N * 4 + 255) & ~(size_t)255;
    float* h = (float*)(ws + off1);                                 // N x 128 (then N x 64)
    size_t off2 = off1 + (((size_t)N * DHID * 4 + 255) & ~(size_t)255);
    float* z = (float*)(ws + off2);                                 // N x 128 (then N x 64)

    // ---- symmetric normalization: dinv = rsqrt(1 + indegree) ----
    k_fill<<<(N + 255) / 256, 256, 0, stream>>>(dinv, 1.0f, N);
    k_deg_accum<<<(nE + 255) / 256, 256, 0, stream>>>(dst, dinv, nE);
    k_rsqrt<<<(N + 255) / 256, 256, 0, stream>>>(dinv, N);

    // ---- layer 1: h = x @ W1 ; z = relu(scatter(h) + b1) ----
    k_gemm_bf16<128><<<(N + 127) / 128, 256, 0, stream>>>(x, W1, h, N, DIN);
    k_self_init<128><<<(N * 128 + 255) / 256, 256, 0, stream>>>(h, dinv, z, N);
    k_scatter<128><<<(nE * 32 + 255) / 256, 256, 0, stream>>>(h, src, dst, dinv, z, nE);
    k_bias_act<128, true><<<(N * 128 + 255) / 256, 256, 0, stream>>>(z, b1, N);

    // ---- layer 2: h = z @ W2 ; z = scatter(h) + b2 ----
    k_gemm_bf16<64><<<(N + 127) / 128, 256, 0, stream>>>(z, W2, h, N, DHID);
    k_self_init<64><<<(N * 64 + 255) / 256, 256, 0, stream>>>(h, dinv, z, N);
    k_scatter<64><<<(nE * 16 + 255) / 256, 256, 0, stream>>>(h, src, dst, dinv, z, nE);
    k_bias_act<64, false><<<(N * 64 + 255) / 256, 256, 0, stream>>>(z, b2, N);

    // ---- decode ----
    k_decode<<<(nL + 255) / 256, 256, 0, stream>>>(z, ls, lt, (float*)d_out, nL, DOUT);

    (void)n_in; (void)out_size; (void)ws_size; (void)DOUT;
}
